// Informer_10436770529862
// MI455X (gfx1250) — matrix operand order repaired and checked by reading, not compiled, not run on hardware
//
#include <hip/hip_runtime.h>
#include <hip/hip_bf16.h>

#define B_   16
#define L_   512
#define CIN  512
#define DM   512
#define NH   8
#define DH   64
#define DFF  2048
#define TKF  51
#define SK   35
#define NTQ  35

typedef __bf16 bf16;
typedef __attribute__((ext_vector_type(16))) __bf16 v16bf;
typedef __attribute__((ext_vector_type(8)))  __bf16 v8bf;
typedef __attribute__((ext_vector_type(8)))  float  v8f;

// ---------------- utility: f32 -> bf16 cast ----------------
__global__ void cast_bf16(const float* __restrict__ in, bf16* __restrict__ out, int n) {
  int i = blockIdx.x * blockDim.x + threadIdx.x;
  if (i < n) out[i] = (bf16)in[i];
}

// ---------------- xm[l] = mean over b,c of x_enc ----------------
__global__ void mean_bc(const float* __restrict__ x, float* __restrict__ xm) {
  __shared__ float red[256];
  int l = blockIdx.x;
  float s = 0.f;
  for (int i = threadIdx.x; i < B_ * CIN; i += 256) {
    int b = i >> 9, c = i & 511;
    s += x[((size_t)b * L_ + l) * CIN + c];
  }
  red[threadIdx.x] = s; __syncthreads();
  for (int o = 128; o > 0; o >>= 1) {
    if (threadIdx.x < o) red[threadIdx.x] += red[threadIdx.x + o];
    __syncthreads();
  }
  if (threadIdx.x == 0) xm[l] = red[0] * (1.f / (B_ * CIN));
}

// ---------------- feature-importance MLP + top-51 select ----------------
__global__ void feat_select(const float* __restrict__ xm,
                            const float* __restrict__ w1, const float* __restrict__ b1,
                            const float* __restrict__ w2, const float* __restrict__ b2,
                            int* __restrict__ sel) {
  __shared__ float xs[CIN], hs[CIN], imp[CIN];
  int j = threadIdx.x;                       // 512 threads
  xs[j] = xm[j];
  __syncthreads();
  float a = b1[j];
  for (int i = 0; i < CIN; i++) a += w1[(size_t)j * CIN + i] * xs[i];
  hs[j] = fmaxf(a, 0.f);
  __syncthreads();
  float c = b2[j];
  for (int i = 0; i < CIN; i++) c += w2[(size_t)j * CIN + i] * hs[i];
  imp[j] = 1.f / (1.f + __expf(-c));         // softmax is monotonic -> rank by imp
  __syncthreads();
  if (j == 0) {
    for (int t = 0; t < TKF; t++) {
      int best = 0; float bv = -1e30f;
      for (int i = 0; i < CIN; i++)
        if (imp[i] > bv) { bv = imp[i]; best = i; }  // strict > == JAX tie rule
      sel[t] = best; imp[best] = -2e30f;
    }
  }
}

// ---------------- gather(sel) + width-3 conv + positional encoding ----------------
__global__ void conv_pe(const float* __restrict__ x_enc, const int* __restrict__ sel,
                        const float* __restrict__ cw,
                        float* __restrict__ x0, bf16* __restrict__ x0b) {
  int bl = blockIdx.x; int b = bl / L_, l = bl % L_;
  int d = threadIdx.x;                       // 512 threads
  __shared__ int   ss[TKF];
  __shared__ float rows[3][TKF];
  if (d < TKF) ss[d] = sel[d];
  __syncthreads();
  if (d < 3 * TKF) {
    int r = d / TKF, c = d % TKF;
    int src = (r == 0) ? (l == 0 ? L_ - 1 : l - 1)
            : (r == 1) ? l
                       : (l == L_ - 1 ? 0 : l + 1);
    rows[r][c] = x_enc[((size_t)b * L_ + src) * CIN + ss[c]];
  }
  __syncthreads();
  float acc = 0.f;
  #pragma unroll
  for (int r = 0; r < 3; r++)
    for (int c = 0; c < TKF; c++)
      acc += rows[r][c] * cw[((size_t)r * TKF + c) * DM + d];
  int i = d >> 1;
  float ang = (float)l * __expf(-(float)(2 * i) * (logf(10000.f) / (float)DM));
  acc += (d & 1) ? __cosf(ang) : __sinf(ang);
  size_t o = (size_t)bl * DM + d;
  x0[o] = acc; x0b[o] = (bf16)acc;
}

// ---------------- WMMA bf16 GEMM: out = act(MxK) @ wgt(NxK)^T (+bias,+resid,relu) ----
// block = 128 threads (4 waves); wave -> 32(M) x 64(N) tile; block -> 128x64.
// Shared 64x32 B tile per K-step is async-staged into LDS (double buffered) with
// GLOBAL_LOAD_ASYNC_TO_LDS_B128 (ASYNCcnt) and consumed via ds_load_b128.
// grid = (N/64, M/128)
__global__ __launch_bounds__(128) void gemm_bf16_wmma(
    const bf16* __restrict__ act, const bf16* __restrict__ wgt,
    const float* __restrict__ bias, const float* __restrict__ resid,
    float* __restrict__ outF, bf16* __restrict__ outB,
    int M, int N, int K, int relu)
{
  __shared__ __align__(16) bf16 bstage[2][64 * 32];   // 2 x 4KB
  int tid  = threadIdx.x;
  int lane = tid & 31;
  int wave = tid >> 5;
  int half = lane >> 4;
  int lm   = lane & 15;
  int tileM = blockIdx.y * 128 + wave * 32;
  int tileN = blockIdx.x * 64;

  // Stage one 64(N) x 32(K) B tile into LDS buffer `buf` for k-offset k0.
  // Chunk q (16B): n = q>>2, khalf = (q>>1)&1, sub = q&1.
  // LDS layout: byte offset q*16, so lane fragment (n, khalf) = 32B contiguous.
  auto stage = [&](int buf, int k0) {
    #pragma unroll
    for (int c = 0; c < 2; c++) {
      int q   = tid + c * 128;
      int n   = q >> 2, kh = (q >> 1) & 1, sub = q & 1;
      const bf16* src = wgt + (size_t)(tileN + n) * K + k0 + kh * 8 + sub * 16;
      unsigned ldsaddr = (unsigned)(size_t)(&bstage[buf][0]) + (unsigned)q * 16u;
      asm volatile("global_load_async_to_lds_b128 %0, %1, off"
                   :: "v"(ldsaddr), "v"((unsigned long long)(size_t)src)
                   : "memory");
    }
  };

  v8f acc[2][4] = {};
  const bf16* arow0 = act + (size_t)(tileM +  0 + lm) * K + half * 8;
  const bf16* arow1 = act + (size_t)(tileM + 16 + lm) * K + half * 8;

  stage(0, 0);
  int it = 0;
  for (int k0 = 0; k0 < K; k0 += 32, ++it) {
    int cur = it & 1;
    asm volatile("s_wait_asynccnt 0x0" ::: "memory");   // staging of `cur` complete
    __syncthreads();                                    // ...for all 4 waves
    if (k0 + 32 < K) stage(cur ^ 1, k0 + 32);

    // A fragments: 16-bit A layout -> two contiguous 8-elem runs per lane
    v8bf a0lo = *(const v8bf*)(arow0 + k0);
    v8bf a0hi = *(const v8bf*)(arow0 + k0 + 16);
    v8bf a1lo = *(const v8bf*)(arow1 + k0);
    v8bf a1hi = *(const v8bf*)(arow1 + k0 + 16);
    v16bf a0, a1;
    #pragma unroll
    for (int i = 0; i < 8; i++) {
      a0[i] = a0lo[i]; a0[8 + i] = a0hi[i];
      a1[i] = a1lo[i]; a1[8 + i] = a1hi[i];
    }
    if (k0 + 32 < K) {                                  // -> global_prefetch_b8
      __builtin_prefetch(arow0 + k0 + 64, 0, 3);
      __builtin_prefetch(arow1 + k0 + 64, 0, 3);
    }
    #pragma unroll
    for (int j = 0; j < 4; j++) {
      // B fragment (32x16 tile j): lane (half,lm) reads 16 contiguous bf16 from LDS
      const bf16* bp = &bstage[cur][((((j * 16 + lm) << 1) | half) << 4)];
      v16bf bfrag = *(const v16bf*)bp;
      acc[0][j] = __builtin_amdgcn_wmma_f32_16x16x32_bf16(false, a0, false, bfrag,
                                                          (short)0, acc[0][j], false, false);
      acc[1][j] = __builtin_amdgcn_wmma_f32_16x16x32_bf16(false, a1, false, bfrag,
                                                          (short)0, acc[1][j], false, false);
    }
  }

  #pragma unroll
  for (int rr = 0; rr < 2; rr++) {
    #pragma unroll
    for (int j = 0; j < 4; j++) {
      int col = tileN + j * 16 + lm;
      float bv = bias ? bias[col] : 0.f;
      #pragma unroll
      for (int r = 0; r < 8; r++) {
        int row = tileM + rr * 16 + half * 8 + r;   // C/D: VGPR r -> M = r + 8*half
        float v = acc[rr][j][r] + bv;
        size_t o = (size_t)row * N + col;
        if (resid) v += resid[o];
        if (relu)  v = fmaxf(v, 0.f);
        if (outF)  outF[o] = v;
        if (outB)  outB[o] = (bf16)v;
      }
    }
  }
}

// ---------------- sampled scores: M[b,h,l] = max_s(q.k_samp) - mean_s(q.k_samp) ----------------
__global__ void prob_score(const float* __restrict__ Q, const float* __restrict__ Kt,
                           const int* __restrict__ idxs, float* __restrict__ Msc) {
  int gid = blockIdx.x;                      // B*H*L blocks, 64 threads
  int l = gid % L_; int bh = gid / L_; int h = bh % NH; int b = bh / NH;
  __shared__ float qs[DH];
  __shared__ float dots[SK];
  int t = threadIdx.x;
  if (t < DH) qs[t] = Q[((size_t)b * L_ + l) * DM + h * DH + t];
  __syncthreads();
  if (t < SK) {
    int ks = idxs[l * SK + t];
    const float* kr = Kt + ((size_t)b * L_ + ks) * DM + h * DH;
    float s = 0.f;
    for (int d = 0; d < DH; d++) s += qs[d] * kr[d];
    dots[t] = s;
  }
  __syncthreads();
  if (t == 0) {
    float mx = -1e30f, sm = 0.f;
    for (int s = 0; s < SK; s++) { mx = fmaxf(mx, dots[s]); sm += dots[s]; }
    Msc[(size_t)bh * L_ + l] = mx - sm * (1.f / SK);
  }
}

// ---------------- top-35 queries per (b,h) ----------------
__global__ void topq(const float* __restrict__ Msc, int* __restrict__ mtop) {
  __shared__ float tmp[L_];
  int bh = blockIdx.x;
  for (int i = threadIdx.x; i < L_; i += blockDim.x) tmp[i] = Msc[(size_t)bh * L_ + i];
  __syncthreads();
  if (threadIdx.x == 0) {
    for (int t = 0; t < NTQ; t++) {
      int best = 0; float bv = -1e30f;
      for (int i = 0; i < L_; i++)
        if (tmp[i] > bv) { bv = tmp[i]; best = i; }
      mtop[bh * NTQ + t] = best; tmp[best] = -2e30f;
    }
  }
}

// ---------------- full attention for the 35 selected queries ----------------
__global__ void attn_sel(const float* __restrict__ Q, const float* __restrict__ Kt,
                         const float* __restrict__ V, const int* __restrict__ mtop,
                         float* __restrict__ upd) {
  int gid = blockIdx.x;                      // B*H*NTQ blocks, 256 threads
  int u = gid % NTQ; int bh = gid / NTQ; int h = bh % NH; int b = bh / NH;
  int lq = mtop[bh * NTQ + u];
  __shared__ float qs[DH];
  __shared__ float sc[L_];
  __shared__ float red[256];
  int t = threadIdx.x;
  if (t < DH) qs[t] = Q[((size_t)b * L_ + lq) * DM + h * DH + t];
  __syncthreads();
  const float scale = 0.125f;                // 1/sqrt(64)
  for (int k = t; k < L_; k += 256) {
    const float* kr = Kt + ((size_t)b * L_ + k) * DM + h * DH;
    float s = 0.f;
    for (int d = 0; d < DH; d++) s += qs[d] * kr[d];
    sc[k] = s * scale;
  }
  __syncthreads();
  red[t] = fmaxf(sc[t], sc[t + 256]); __syncthreads();
  for (int o = 128; o > 0; o >>= 1) { if (t < o) red[t] = fmaxf(red[t], red[t + o]); __syncthreads(); }
  float mx = red[0]; __syncthreads();
  float e0 = __expf(sc[t] - mx), e1 = __expf(sc[t + 256] - mx);
  sc[t] = e0; sc[t + 256] = e1;
  red[t] = e0 + e1; __syncthreads();
  for (int o = 128; o > 0; o >>= 1) { if (t < o) red[t] += red[t + o]; __syncthreads(); }
  float inv = 1.f / red[0]; __syncthreads();
  if (t < DH) {
    float s = 0.f;
    for (int k = 0; k < L_; k++) s += sc[k] * V[((size_t)b * L_ + k) * DM + h * DH + t];
    upd[((size_t)gid) * DH + t] = s * inv;
  }
}

// ---------------- v mean over L per (b,h,d) ----------------
__global__ void v_mean(const float* __restrict__ V, float* __restrict__ vm) {
  int bh = blockIdx.x; int h = bh % NH; int b = bh / NH;
  int d = threadIdx.x;                       // 64 threads
  float s = 0.f;
  for (int l = 0; l < L_; l++) s += V[((size_t)b * L_ + l) * DM + h * DH + d];
  vm[bh * DH + d] = s * (1.f / L_);
}

// ---------------- ctx fill (broadcast v-mean) then scatter upd rows ----------------
__global__ void ctx_fill(const float* __restrict__ vm, bf16* __restrict__ ctxb) {
  int bl = blockIdx.x; int b = bl / L_;
  int d = threadIdx.x;                       // 512 threads
  int h = d >> 6;
  ctxb[(size_t)bl * DM + d] = (bf16)vm[(b * NH + h) * DH + (d & 63)];
}

__global__ void ctx_scatter(const float* __restrict__ upd, const int* __restrict__ mtop,
                            bf16* __restrict__ ctxb) {
  int gid = blockIdx.x; int u = gid % NTQ; int bh = gid / NTQ;
  int h = bh % NH; int b = bh / NH;
  int lq = mtop[bh * NTQ + u];
  int d = threadIdx.x;                       // 64 threads
  ctxb[((size_t)b * L_ + lq) * DM + h * DH + d] = (bf16)upd[((size_t)gid) * DH + d];
}

// ---------------- layernorm over last dim (512), optional bf16 mirror ----------------
__global__ void layernorm(const float* __restrict__ x, const float* __restrict__ g,
                          const float* __restrict__ bta,
                          float* __restrict__ outF, bf16* __restrict__ outB) {
  __shared__ float red[256], red2[256];
  int row = blockIdx.x; int t = threadIdx.x; // 256 threads
  const float* xr = x + (size_t)row * DM;
  float a0 = xr[t], a1 = xr[t + 256];
  red[t] = a0 + a1; red2[t] = a0 * a0 + a1 * a1;
  __syncthreads();
  for (int o = 128; o > 0; o >>= 1) {
    if (t < o) { red[t] += red[t + o]; red2[t] += red2[t + o]; }
    __syncthreads();
  }
  float mean = red[0] * (1.f / DM);
  float var  = red2[0] * (1.f / DM) - mean * mean;
  float inv  = rsqrtf(var + 1e-5f);
  float y0 = (a0 - mean) * inv * g[t]       + bta[t];
  float y1 = (a1 - mean) * inv * g[t + 256] + bta[t + 256];
  size_t o0 = (size_t)row * DM + t, o1 = o0 + 256;
  if (outF) { outF[o0] = y0; outF[o1] = y1; }
  if (outB) { outB[o0] = (bf16)y0; outB[o1] = (bf16)y1; }
}

extern "C" void kernel_launch(void* const* d_in, const int* in_sizes, int n_in,
                              void* d_out, int out_size, void* d_ws, size_t ws_size,
                              hipStream_t stream) {
  (void)in_sizes; (void)n_in; (void)out_size; (void)ws_size;
  const float* x_enc  = (const float*)d_in[0];
  const float* fi_w1  = (const float*)d_in[1];
  const float* fi_b1  = (const float*)d_in[2];
  const float* fi_w2  = (const float*)d_in[3];
  const float* fi_b2  = (const float*)d_in[4];
  const float* conv_w = (const float*)d_in[5];
  const float* Wq = (const float*)d_in[6];  const float* bq = (const float*)d_in[7];
  const float* Wk = (const float*)d_in[8];  const float* bk = (const float*)d_in[9];
  const float* Wv = (const float*)d_in[10]; const float* bv = (const float*)d_in[11];
  const float* Wo = (const float*)d_in[12]; const float* bo = (const float*)d_in[13];
  const float* ffn_w1 = (const float*)d_in[14]; const float* ffn_b1 = (const float*)d_in[15];
  const float* ffn_w2 = (const float*)d_in[16]; const float* ffn_b2 = (const float*)d_in[17];
  const float* ln1_g = (const float*)d_in[18]; const float* ln1_b = (const float*)d_in[19];
  const float* ln2_g = (const float*)d_in[20]; const float* ln2_b = (const float*)d_in[21];
  const int*   index_sample = (const int*)d_in[22];

  const int ML = B_ * L_;                    // 8192 rows
  char* ws = (char*)d_ws;
  size_t off = 0;
  auto carve = [&](size_t bytes) -> char* {
    char* p = ws + off; off += (bytes + 255) & ~(size_t)255; return p;
  };
  float* xm   = (float*)carve(512 * 4);
  int*   sel  = (int*)  carve(64 * 4);
  float* x0   = (float*)carve((size_t)ML * DM * 4);
  bf16*  x0b  = (bf16*) carve((size_t)ML * DM * 2);
  float* Qb   = (float*)carve((size_t)ML * DM * 4);
  float* Kb   = (float*)carve((size_t)ML * DM * 4);
  float* Vb   = (float*)carve((size_t)ML * DM * 4);
  float* Msc  = (float*)carve((size_t)B_ * NH * L_ * 4);
  int*   mtop = (int*)  carve((size_t)B_ * NH * NTQ * 4);
  float* updb = (float*)carve((size_t)B_ * NH * NTQ * DH * 4);
  float* vm   = (float*)carve((size_t)B_ * NH * DH * 4);
  bf16*  ctxb = (bf16*) carve((size_t)ML * DM * 2);
  float* x1   = (float*)carve((size_t)ML * DM * 4);
  float* x1n  = (float*)carve((size_t)ML * DM * 4);
  bf16*  x1nb = (bf16*) carve((size_t)ML * DM * 2);
  bf16*  y1b  = (bf16*) carve((size_t)ML * DFF * 2);
  float* y2   = (float*)carve((size_t)ML * DM * 4);
  bf16*  wqb  = (bf16*) carve((size_t)DM * DM * 2);
  bf16*  wkb  = (bf16*) carve((size_t)DM * DM * 2);
  bf16*  wvb  = (bf16*) carve((size_t)DM * DM * 2);
  bf16*  wob  = (bf16*) carve((size_t)DM * DM * 2);
  bf16*  w1b  = (bf16*) carve((size_t)DFF * DM * 2);
  bf16*  w2b  = (bf16*) carve((size_t)DM * DFF * 2);

  // weight bf16 casts
  int nw = DM * DM;
  cast_bf16<<<(nw + 255) / 256, 256, 0, stream>>>(Wq, wqb, nw);
  cast_bf16<<<(nw + 255) / 256, 256, 0, stream>>>(Wk, wkb, nw);
  cast_bf16<<<(nw + 255) / 256, 256, 0, stream>>>(Wv, wvb, nw);
  cast_bf16<<<(nw + 255) / 256, 256, 0, stream>>>(Wo, wob, nw);
  int nf = DFF * DM;
  cast_bf16<<<(nf + 255) / 256, 256, 0, stream>>>(ffn_w1, w1b, nf);
  cast_bf16<<<(nf + 255) / 256, 256, 0, stream>>>(ffn_w2, w2b, nf);

  // feature selection + conv + PE
  mean_bc    <<<L_, 256, 0, stream>>>(x_enc, xm);
  feat_select<<<1, 512, 0, stream>>>(xm, fi_w1, fi_b1, fi_w2, fi_b2, sel);
  conv_pe    <<<ML, 512, 0, stream>>>(x_enc, sel, conv_w, x0, x0b);

  // QKV projections (WMMA, async-LDS staged B)
  dim3 g512(DM / 64, ML / 128);              // (8,64)
  gemm_bf16_wmma<<<g512, 128, 0, stream>>>(x0b, wqb, bq, nullptr, Qb, nullptr, ML, DM, DM, 0);
  gemm_bf16_wmma<<<g512, 128, 0, stream>>>(x0b, wkb, bk, nullptr, Kb, nullptr, ML, DM, DM, 0);
  gemm_bf16_wmma<<<g512, 128, 0, stream>>>(x0b, wvb, bv, nullptr, Vb, nullptr, ML, DM, DM, 0);

  // ProbSparse attention
  prob_score <<<B_ * NH * L_, 64, 0, stream>>>(Qb, Kb, index_sample, Msc);
  topq       <<<B_ * NH, 64, 0, stream>>>(Msc, mtop);
  attn_sel   <<<B_ * NH * NTQ, 256, 0, stream>>>(Qb, Kb, Vb, mtop, updb);
  v_mean     <<<B_ * NH, 64, 0, stream>>>(Vb, vm);
  ctx_fill   <<<ML, 512, 0, stream>>>(vm, ctxb);
  ctx_scatter<<<B_ * NH * NTQ, 64, 0, stream>>>(updb, mtop, ctxb);

  // output projection + residual, LN1
  gemm_bf16_wmma<<<g512, 128, 0, stream>>>(ctxb, wob, bo, x0, x1, nullptr, ML, DM, DM, 0);
  layernorm <<<ML, 256, 0, stream>>>(x1, ln1_g, ln1_b, x1n, x1nb);

  // FFN (WMMA), residual, LN2 -> d_out
  dim3 gff1(DFF / 64, ML / 128);             // (32,64)
  gemm_bf16_wmma<<<gff1, 128, 0, stream>>>(x1nb, w1b, ffn_b1, nullptr, nullptr, y1b, ML, DFF, DM, 1);
  gemm_bf16_wmma<<<g512, 128, 0, stream>>>(y1b, w2b, ffn_b2, x1n, y2, nullptr, ML, DM, DFF, 0);
  layernorm <<<ML, 256, 0, stream>>>(y2, ln2_g, ln2_b, (float*)d_out, nullptr);
}
